// Encoder_37915971289796
// MI455X (gfx1250) — matrix operand order repaired and checked by reading, not compiled, not run on hardware
//
#include <hip/hip_runtime.h>

typedef __attribute__((ext_vector_type(16))) __bf16 v16bf;
typedef __attribute__((ext_vector_type(8)))  float  v8f;
typedef int v4i __attribute__((ext_vector_type(4)));

#define WMMA_BF16(A, B, C) \
  __builtin_amdgcn_wmma_f32_16x16x32_bf16(false, (A), false, (B), (short)0, (C), false, false)

// gfx1250 async global->LDS path (ASYNCcnt-tracked). Probe result from last
// round: builtin exists with signature (AS1 int4*, AS3 int4*, imm, imm).
#if defined(__has_builtin)
#if __has_builtin(__builtin_amdgcn_global_load_async_to_lds_b128)
#define HAVE_ASYNC 1
#endif
#endif

#if defined(HAVE_ASYNC)
typedef __attribute__((address_space(1))) void* as1_vp;
typedef __attribute__((address_space(3))) void* as3_vp;
typedef __attribute__((address_space(1))) v4i*  gas_ptr;
typedef __attribute__((address_space(3))) v4i*  las_ptr;
#define ASYNC_B128(g, l, off) \
  __builtin_amdgcn_global_load_async_to_lds_b128( \
      (gas_ptr)(as1_vp)(void*)(g), (las_ptr)(as3_vp)(void*)(l), (off), 0)
#if __has_builtin(__builtin_amdgcn_s_wait_asynccnt)
#define WAIT_ASYNC() __builtin_amdgcn_s_wait_asynccnt(0)
#else
#define WAIT_ASYNC() asm volatile("s_wait_asynccnt 0" ::: "memory")
#endif
#endif

// ---------------------------------------------------------------- constants
static constexpr int L  = 6;
static constexpr int D  = 1024;
static constexpr int H  = 16;
static constexpr int F  = 4096;
static constexpr int S  = 512;
static constexpr int B  = 16;
static constexpr int DH = 64;
static constexpr int M  = B * S;          // 8192 rows
static constexpr float SCALE = 0.125f;    // 1/sqrt(64)

// ---------------------------------------------------------------- h = x + pe
__global__ __launch_bounds__(256) void addpe_k(const float* __restrict__ x,
                                               const float* __restrict__ pe,
                                               float* __restrict__ h, int n) {
  int i = blockIdx.x * 256 + threadIdx.x;
  if (i < n) h[i] = x[i] + pe[i & (S * D - 1)];   // S*D = 2^19
}

// ----------------------------------------- LDS-tiled W[k][n] f32 -> WT[n][k] bf16
// 64x64 tile per block; both global read and write are >=32B contiguous/lane.
__global__ __launch_bounds__(256) void wconv_k(const float* __restrict__ in,
                                               __bf16* __restrict__ out,
                                               int K, int N) {
  __shared__ __bf16 st[64][72];
  const int n0 = blockIdx.x * 64;
  const int k0 = blockIdx.y * 64;
  const int r = threadIdx.x >> 2;          // 0..63
  const int cg = threadIdx.x & 3;          // 4 x 16-element column groups
  const float* ip = in + (size_t)(k0 + r) * N + n0 + cg * 16;
#pragma unroll
  for (int i = 0; i < 16; i++) st[r][cg * 16 + i] = (__bf16)ip[i];
  __syncthreads();
  __bf16* op = out + (size_t)(n0 + r) * K + k0 + cg * 16;
#pragma unroll
  for (int i = 0; i < 16; i++) op[i] = st[cg * 16 + i][r];
}

// -------------------------------------------------------- layernorm -> bf16
__global__ __launch_bounds__(128) void ln_k(const float* __restrict__ x,
                                            const float* __restrict__ g,
                                            const float* __restrict__ bb,
                                            __bf16* __restrict__ out) {
  const int row = blockIdx.x;
  const int t = threadIdx.x;
  const float* xr = x + (size_t)row * D;
  float v[8];
  float s = 0.f;
#pragma unroll
  for (int i = 0; i < 8; i++) { v[i] = xr[t + i * 128]; s += v[i]; }
#pragma unroll
  for (int o = 16; o >= 1; o >>= 1) s += __shfl_xor(s, o, 32);
  __shared__ float red[4], red2[4];
  if ((t & 31) == 0) red[t >> 5] = s;
  __syncthreads();
  s = red[0] + red[1] + red[2] + red[3];
  float mu = s * (1.0f / D);
  float vs = 0.f;
#pragma unroll
  for (int i = 0; i < 8; i++) { float d = v[i] - mu; vs += d * d; }
#pragma unroll
  for (int o = 16; o >= 1; o >>= 1) vs += __shfl_xor(vs, o, 32);
  if ((t & 31) == 0) red2[t >> 5] = vs;
  __syncthreads();
  vs = red2[0] + red2[1] + red2[2] + red2[3];
  float rstd = rsqrtf(vs * (1.0f / D) + 1e-6f);
#pragma unroll
  for (int i = 0; i < 8; i++) {
    int c = t + i * 128;
    out[(size_t)row * D + c] = (__bf16)((v[i] - mu) * rstd * g[c] + bb[c]);
  }
}

// ---------------------------------------------------------------- WMMA GEMM
// C = A[bf16 M x K] * BT[bf16 N x K]^T + bias. Block 128x128, 8 waves in a
// 2(M) x 4(N) grid, each wave owns 64x32 (4x2 WMMA tiles). Double-buffered
// LDS; async global->LDS when available, else register-staged.
// MODE 0: Q/K head layout  [B,H,S,DH] bf16
// MODE 1: V transposed     [B,H,DH,S] bf16
// MODE 2: out1 = acc + bias + res1                (f32)
// MODE 3: gelu(acc + bias)                        (bf16)
// MODE 4: h    = acc + bias + res1 + res2         (f32, in-place over res2)
static constexpr int BM = 128, BN = 128, BK = 32, ASTR = 40;

template <int MODE>
__global__ __launch_bounds__(256) void gemm_k(const __bf16* __restrict__ A,
                                              const __bf16* __restrict__ BT,
                                              const float* __restrict__ bias,
                                              __bf16* __restrict__ obf,
                                              float* __restrict__ of32,
                                              const float* __restrict__ res1,
                                              const float* __restrict__ res2,
                                              int N, int K) {
  __shared__ __align__(16) __bf16 sA[2][BM * ASTR];
  __shared__ __align__(16) __bf16 sB[2][BN * ASTR];
  const int tid = threadIdx.x;
  const int lane = tid & 31, wave = tid >> 5;
  const int laneLo = lane & 15, laneHi = lane >> 4;
  const int mw = wave & 1, nw = wave >> 1;        // 2 x 4 wave grid
  const int blockN = blockIdx.x * BN;
  const int blockM = blockIdx.y * BM;
  const int row = tid >> 1, half = tid & 1;       // 128 rows x 2 halves of 16
  const __bf16* ag = A  + (size_t)(blockM + row) * K + half * 16;
  const __bf16* bg = BT + (size_t)(blockN + row) * K + half * 16;
  const int ldsOff = row * ASTR + half * 16;

  v8f acc[4][2] = {};
  const int nk = K / BK;

#if defined(HAVE_ASYNC)
  // stage 0 straight into LDS, no VGPR staging
  ASYNC_B128(ag, &sA[0][ldsOff], 0);
  ASYNC_B128(ag, &sA[0][ldsOff], 16);
  ASYNC_B128(bg, &sB[0][ldsOff], 0);
  ASYNC_B128(bg, &sB[0][ldsOff], 16);
  WAIT_ASYNC();
  __syncthreads();
#else
  {
    uint4 a0 = *(const uint4*)ag, a1 = *(const uint4*)(ag + 8);
    uint4 b0 = *(const uint4*)bg, b1 = *(const uint4*)(bg + 8);
    *(uint4*)&sA[0][ldsOff] = a0; *(uint4*)&sA[0][ldsOff + 8] = a1;
    *(uint4*)&sB[0][ldsOff] = b0; *(uint4*)&sB[0][ldsOff + 8] = b1;
  }
  __syncthreads();
#endif

  for (int kt = 0; kt < nk; kt++) {
    const int cur = kt & 1, nxt = cur ^ 1;
    const bool has_next = (kt + 1 < nk);
#if defined(HAVE_ASYNC)
    if (has_next) {
      const __bf16* an = ag + (kt + 1) * BK;
      const __bf16* bn = bg + (kt + 1) * BK;
      ASYNC_B128(an, &sA[nxt][ldsOff], 0);
      ASYNC_B128(an, &sA[nxt][ldsOff], 16);
      ASYNC_B128(bn, &sB[nxt][ldsOff], 0);
      ASYNC_B128(bn, &sB[nxt][ldsOff], 16);
      if (kt + 2 < nk) {
        __builtin_prefetch(ag + (kt + 2) * BK, 0, 1);
        __builtin_prefetch(bg + (kt + 2) * BK, 0, 1);
      }
    }
#else
    uint4 a0, a1, b0, b1;
    if (has_next) {
      const __bf16* an = ag + (kt + 1) * BK;
      const __bf16* bn = bg + (kt + 1) * BK;
      a0 = *(const uint4*)an; a1 = *(const uint4*)(an + 8);
      b0 = *(const uint4*)bn; b1 = *(const uint4*)(bn + 8);
      if (kt + 2 < nk) {
        __builtin_prefetch(ag + (kt + 2) * BK, 0, 1);
        __builtin_prefetch(bg + (kt + 2) * BK, 0, 1);
      }
    }
#endif

    v16bf af[4], bf[2];
#pragma unroll
    for (int i = 0; i < 4; i++) {
      const __bf16* p = &sA[cur][(mw * 64 + i * 16 + laneLo) * ASTR + laneHi * 16];
      ((uint4*)&af[i])[0] = *(const uint4*)p;
      ((uint4*)&af[i])[1] = *(const uint4*)(p + 8);
    }
#pragma unroll
    for (int j = 0; j < 2; j++) {
      const __bf16* p = &sB[cur][(nw * 32 + j * 16 + laneLo) * ASTR + laneHi * 16];
      ((uint4*)&bf[j])[0] = *(const uint4*)p;
      ((uint4*)&bf[j])[1] = *(const uint4*)(p + 8);
    }
#pragma unroll
    for (int i = 0; i < 4; i++)
#pragma unroll
      for (int j = 0; j < 2; j++)
        acc[i][j] = WMMA_BF16(af[i], bf[j], acc[i][j]);

#if defined(HAVE_ASYNC)
    if (has_next) WAIT_ASYNC();
#else
    if (has_next) {
      *(uint4*)&sA[nxt][ldsOff] = a0; *(uint4*)&sA[nxt][ldsOff + 8] = a1;
      *(uint4*)&sB[nxt][ldsOff] = b0; *(uint4*)&sB[nxt][ldsOff + 8] = b1;
    }
#endif
    __syncthreads();
  }

#pragma unroll
  for (int i = 0; i < 4; i++)
#pragma unroll
    for (int j = 0; j < 2; j++)
#pragma unroll
      for (int l = 0; l < 8; l++) {
        int m = blockM + mw * 64 + i * 16 + l + 8 * laneHi;
        int n = blockN + nw * 32 + j * 16 + laneLo;
        float v = acc[i][j][l] + bias[n];
        size_t mn = (size_t)m * N + n;
        if (MODE == 0) {
          int bb = m >> 9, ss = m & 511, hh = n >> 6, dd = n & 63;
          obf[(((size_t)(bb * H + hh) * S) + ss) * DH + dd] = (__bf16)v;
        } else if (MODE == 1) {
          int bb = m >> 9, ss = m & 511, hh = n >> 6, dd = n & 63;
          obf[(((size_t)(bb * H + hh) * DH) + dd) * S + ss] = (__bf16)v;
        } else if (MODE == 2) {
          of32[mn] = v + res1[mn];
        } else if (MODE == 3) {
          obf[mn] = (__bf16)(0.5f * v * (1.0f + erff(v * 0.70710678118f)));
        } else {
          of32[mn] = v + res1[mn] + res2[mn];
        }
      }
}

// ------------------------------------------------------- fused attention
// One block = (b, head, 16 queries). 4 waves / 128 threads.
static constexpr int SSTR = 520;   // padded row stride (conflict-free 16B lanes)

__global__ __launch_bounds__(128) void attn_k(const __bf16* __restrict__ Q,
                                              const __bf16* __restrict__ Kb,
                                              const __bf16* __restrict__ Vt,
                                              const float* __restrict__ btab,
                                              __bf16* __restrict__ ctx) {
  __shared__ __align__(16) float  sS[16 * SSTR];
  __shared__ __align__(16) __bf16 sP[16 * SSTR];
  const int blk = blockIdx.x;
  const int qb = blk & 31;          // S/16 = 32 query blocks
  const int bh = blk >> 5;          // b*H + head
  const int head = bh & 15;
  const int q0 = qb * 16;
  const int tid = threadIdx.x;
  const int lane = tid & 31, wave = tid >> 5;
  const int laneLo = lane & 15, laneHi = lane >> 4;
  const __bf16* Qp = Q  + (size_t)bh * S * DH;
  const __bf16* Kp = Kb + (size_t)bh * S * DH;
  const __bf16* Vp = Vt + (size_t)bh * DH * S;

  // ---- phase 1: scores = scale * Q K^T + rel-pos bias -> LDS f32
  v16bf aq[2];
#pragma unroll
  for (int ks = 0; ks < 2; ks++) {
    const __bf16* p = Qp + (size_t)(q0 + laneLo) * DH + ks * 32 + laneHi * 16;
    ((uint4*)&aq[ks])[0] = *(const uint4*)p;
    ((uint4*)&aq[ks])[1] = *(const uint4*)(p + 8);
  }
  for (int nt = wave; nt < 32; nt += 4) {
    v8f acc = {};
#pragma unroll
    for (int ks = 0; ks < 2; ks++) {
      v16bf bk;
      const __bf16* p = Kp + (size_t)(nt * 16 + laneLo) * DH + ks * 32 + laneHi * 16;
      ((uint4*)&bk)[0] = *(const uint4*)p;
      ((uint4*)&bk)[1] = *(const uint4*)(p + 8);
      acc = WMMA_BF16(aq[ks], bk, acc);
    }
#pragma unroll
    for (int l = 0; l < 8; l++) {
      int qloc = l + 8 * laneHi;
      int kg = nt * 16 + laneLo;
      int rel = (q0 + qloc) - kg + 511;                // [0, 1022]
      sS[qloc * SSTR + kg] = acc[l] * SCALE + btab[rel * H + head];
    }
  }
  __syncthreads();

  // ---- phase 2: softmax (8 threads per row) -> bf16 probs in LDS
  {
    int row = tid >> 3, sub = tid & 7;
    float* rp = sS + row * SSTR;
    float mx = -3.0e38f;
#pragma unroll
    for (int i = 0; i < 64; i++) mx = fmaxf(mx, rp[sub + i * 8]);
    mx = fmaxf(mx, __shfl_xor(mx, 1, 32));
    mx = fmaxf(mx, __shfl_xor(mx, 2, 32));
    mx = fmaxf(mx, __shfl_xor(mx, 4, 32));
    float e[64];
    float sum = 0.f;
#pragma unroll
    for (int i = 0; i < 64; i++) { e[i] = __expf(rp[sub + i * 8] - mx); sum += e[i]; }
    sum += __shfl_xor(sum, 1, 32);
    sum += __shfl_xor(sum, 2, 32);
    sum += __shfl_xor(sum, 4, 32);
    float inv = 1.0f / sum;
    __bf16* pp = sP + row * SSTR;
#pragma unroll
    for (int i = 0; i < 64; i++) pp[sub + i * 8] = (__bf16)(e[i] * inv);
  }
  __syncthreads();

  // ---- phase 3: ctx = P @ V  (wave w owns d-tile w; K = 512 keys)
  {
    v8f acc = {};
    for (int ks = 0; ks < 16; ks++) {
      v16bf a, bv;
      const __bf16* pa = sP + laneLo * SSTR + ks * 32 + laneHi * 16;
      ((uint4*)&a)[0] = *(const uint4*)pa;
      ((uint4*)&a)[1] = *(const uint4*)(pa + 8);
      const __bf16* pb = Vp + (size_t)(wave * 16 + laneLo) * S + ks * 32 + laneHi * 16;
      ((uint4*)&bv)[0] = *(const uint4*)pb;
      ((uint4*)&bv)[1] = *(const uint4*)(pb + 8);
      acc = WMMA_BF16(a, bv, acc);
    }
    int bb = bh >> 4;
#pragma unroll
    for (int l = 0; l < 8; l++) {
      int qg = q0 + l + 8 * laneHi;
      int dd = wave * 16 + laneLo;
      ctx[((size_t)(bb * S + qg)) * D + head * DH + dd] = (__bf16)acc[l];
    }
  }
}

// ---------------------------------------------------------------- launcher
extern "C" void kernel_launch(void* const* d_in, const int* in_sizes, int n_in,
                              void* d_out, int out_size, void* d_ws, size_t ws_size,
                              hipStream_t stream) {
  (void)in_sizes; (void)n_in; (void)out_size; (void)ws_size;
  const float* x    = (const float*)d_in[0];
  const float* pe   = (const float*)d_in[1];
  const float* wq   = (const float*)d_in[2];
  const float* bq   = (const float*)d_in[3];
  const float* wk   = (const float*)d_in[4];
  const float* bk   = (const float*)d_in[5];
  const float* wv   = (const float*)d_in[6];
  const float* bv   = (const float*)d_in[7];
  const float* wo   = (const float*)d_in[8];
  const float* bo   = (const float*)d_in[9];
  const float* btab = (const float*)d_in[10];
  const float* w1   = (const float*)d_in[11];
  const float* b1   = (const float*)d_in[12];
  const float* w2   = (const float*)d_in[13];
  const float* b2   = (const float*)d_in[14];
  const float* g1   = (const float*)d_in[15];
  const float* be1  = (const float*)d_in[16];
  const float* g2   = (const float*)d_in[17];
  const float* be2  = (const float*)d_in[18];

  char* ws = (char*)d_ws;
  auto carve = [&](size_t bytes) { char* p = ws; ws += (bytes + 255) & ~(size_t)255; return p; };
  float*  h    = (float*)carve((size_t)M * D * 4);
  float*  out1 = (float*)carve((size_t)M * D * 4);
  __bf16* xn   = (__bf16*)carve((size_t)M * D * 2);
  __bf16* qb_  = (__bf16*)carve((size_t)M * D * 2);
  __bf16* kb_  = (__bf16*)carve((size_t)M * D * 2);
  __bf16* vb_  = (__bf16*)carve((size_t)M * D * 2);
  __bf16* ctx  = (__bf16*)carve((size_t)M * D * 2);
  __bf16* ffn  = (__bf16*)carve((size_t)M * F * 2);
  __bf16* wqT  = (__bf16*)carve((size_t)D * D * 2);
  __bf16* wkT  = (__bf16*)carve((size_t)D * D * 2);
  __bf16* wvT  = (__bf16*)carve((size_t)D * D * 2);
  __bf16* woT  = (__bf16*)carve((size_t)D * D * 2);
  __bf16* w1T  = (__bf16*)carve((size_t)D * F * 2);
  __bf16* w2T  = (__bf16*)carve((size_t)F * D * 2);

  addpe_k<<<(M * D + 255) / 256, 256, 0, stream>>>(x, pe, h, M * D);

  dim3 blk(256);
  dim3 gDD(D / 64, D / 64);        // transpose grids (N/64, K/64)
  dim3 gW1(F / 64, D / 64);
  dim3 gW2(D / 64, F / 64);
  dim3 gridP(D / BN, M / BM);      // N=1024 GEMMs
  dim3 gridF(F / BN, M / BM);      // N=4096 GEMM

  for (int i = 0; i < L; i++) {
    // per-layer weight convert+transpose to bf16 [N][K]
    wconv_k<<<gDD, blk, 0, stream>>>(wq + (size_t)i * D * D, wqT, D, D);
    wconv_k<<<gDD, blk, 0, stream>>>(wk + (size_t)i * D * D, wkT, D, D);
    wconv_k<<<gDD, blk, 0, stream>>>(wv + (size_t)i * D * D, wvT, D, D);
    wconv_k<<<gDD, blk, 0, stream>>>(wo + (size_t)i * D * D, woT, D, D);
    wconv_k<<<gW1, blk, 0, stream>>>(w1 + (size_t)i * D * F, w1T, D, F);
    wconv_k<<<gW2, blk, 0, stream>>>(w2 + (size_t)i * F * D, w2T, F, D);

    // ln1 -> xn (bf16)
    ln_k<<<M, 128, 0, stream>>>(h, g1 + i * D, be1 + i * D, xn);

    // Q, K, V projections
    gemm_k<0><<<gridP, blk, 0, stream>>>(xn, wqT, bq + i * D, qb_, nullptr, nullptr, nullptr, D, D);
    gemm_k<0><<<gridP, blk, 0, stream>>>(xn, wkT, bk + i * D, kb_, nullptr, nullptr, nullptr, D, D);
    gemm_k<1><<<gridP, blk, 0, stream>>>(xn, wvT, bv + i * D, vb_, nullptr, nullptr, nullptr, D, D);

    // fused attention
    attn_k<<<B * H * (S / 16), 128, 0, stream>>>(qb_, kb_, vb_, btab + (size_t)i * (2 * S - 1) * H, ctx);

    // out projection + residual: out1 = h + ctx@wo + bo
    gemm_k<2><<<gridP, blk, 0, stream>>>(ctx, woT, bo + i * D, nullptr, out1, h, nullptr, D, D);

    // ln2 -> xn (bf16)
    ln_k<<<M, 128, 0, stream>>>(out1, g2 + i * D, be2 + i * D, xn);

    // FFN up + gelu
    gemm_k<3><<<gridF, blk, 0, stream>>>(xn, w1T, b1 + i * F, ffn, nullptr, nullptr, nullptr, F, D);

    // FFN down + double residual: h = out1 + ffn@w2 + b2 + h   (in place)
    gemm_k<4><<<gridP, blk, 0, stream>>>(ffn, w2T, b2 + i * D, nullptr, h, out1, h, D, F);
  }

  (void)hipMemcpyAsync(d_out, h, (size_t)M * D * 4, hipMemcpyDeviceToDevice, stream);
}
